// TimelineGNNLayer7_39410619908408
// MI455X (gfx1250) — matrix operand order repaired and checked by reading, not compiled, not run on hardware
//
#include <hip/hip_runtime.h>
#include <hip/hip_bf16.h>

// ---------------------------------------------------------------------------
// Types for CDNA5 WMMA (wave32): A/B = 16 bf16 per lane, C/D = 8 f32 per lane
// ---------------------------------------------------------------------------
typedef __bf16 bf16_t;
typedef bf16_t v16bf __attribute__((ext_vector_type(16)));
typedef bf16_t v8bf  __attribute__((ext_vector_type(8)));
typedef bf16_t v4bf  __attribute__((ext_vector_type(4)));
typedef float  v8f   __attribute__((ext_vector_type(8)));

union V16U { v16bf v; v8bf h[2]; };

#define PER_WAVE_LDS 77952   // bytes of LDS staging per wave (4 waves/block)

__device__ __forceinline__ float fast_sigmoid(float x) {
  return __fdividef(1.0f, 1.0f + __expf(-x));
}
__device__ __forceinline__ float fast_tanh(float x) {
  return 2.0f * fast_sigmoid(2.0f * x) - 1.0f;
}
__device__ __forceinline__ float leaky(float x) {
  return x > 0.0f ? x : 0.01f * x;
}

// CDNA5 async copy: global memory -> LDS, 16B per active lane, ASYNCcnt.
__device__ __forceinline__ void async_copy_b128(void* lds_dst, const void* gsrc) {
  // Generic LDS pointers keep the LDS byte offset in the low 32 bits.
  unsigned lds = (unsigned)(unsigned long long)lds_dst;
  asm volatile("global_load_async_to_lds_b128 %0, %1, off"
               :: "v"(lds), "v"(gsrc)
               : "memory");
}
__device__ __forceinline__ void wait_async0() {
  asm volatile("s_wait_asynccnt 0x0" ::: "memory");
}

__device__ __forceinline__ void wmma_bf16(const v16bf& a, const v16bf& b, v8f& c) {
  c = __builtin_amdgcn_wmma_f32_16x16x32_bf16(false, a, false, b, (short)0, c,
                                              false, false);
}

// ---------------------------------------------------------------------------
// D(32x16 tile) += A(32 x 32*NKS, LDS row-major bf16) @ B(fragment-swizzled)
// A layout per CDNA5 ISA: lane m=(l&15), K-half g=(l>>4); per k-step the lane
// reads k = [8g .. 8g+7] and [16+8g .. 23+8g]  -> two 16-byte LDS loads.
// B fragments are pre-swizzled: frag f = (ntile*ksTotal + ks), lane-contiguous
// 32 bytes per lane -> two 16-byte global loads (hot in L2).
// ---------------------------------------------------------------------------
template <int NKS>
__device__ __forceinline__ void gemm_acc(const bf16_t* __restrict__ aBase,
                                         int aStride,
                                         const bf16_t* __restrict__ wBase,
                                         int fragBase, int lane, v8f& c0,
                                         v8f& c1) {
  const int mrow = lane & 15;
  const int g = lane >> 4;
  const bf16_t* a0 = aBase + mrow * aStride;
  const bf16_t* a1 = aBase + (mrow + 16) * aStride;
#pragma unroll
  for (int ks = 0; ks < NKS; ++ks) {
    const v8bf* bp =
        (const v8bf*)(wBase + (size_t)(fragBase + ks) * 512 + lane * 16);
    V16U B;
    B.h[0] = bp[0];
    B.h[1] = bp[1];
    const int k0 = ks * 32 + 8 * g;
    V16U A;
    A.h[0] = *(const v8bf*)(a0 + k0);
    A.h[1] = *(const v8bf*)(a0 + k0 + 16);
    wmma_bf16(A.v, B.v, c0);
    V16U A2;
    A2.h[0] = *(const v8bf*)(a1 + k0);
    A2.h[1] = *(const v8bf*)(a1 + k0 + 16);
    wmma_bf16(A2.v, B.v, c1);
  }
}

__device__ __forceinline__ v4bf pack4(float a, float b, float c, float d) {
  v4bf p;
  p.x = (bf16_t)a;
  p.y = (bf16_t)b;
  p.z = (bf16_t)c;
  p.w = (bf16_t)d;
  return p;
}

// ---------------------------------------------------------------------------
// Weight prep: fp32 row-major (K x N)  ->  bf16 WMMA-B fragment layout
// dst[f*512 + lane*16 + j] = W[(ks*32 + (lane>>4)*8 + (j&7) + (j>>3)*16) * N + nt*16+(lane&15)]
// with f = nt*(K/32) + ks
// ---------------------------------------------------------------------------
__global__ void swizzle_kernel(const float* __restrict__ src,
                               bf16_t* __restrict__ dst, int K, int N) {
  int idx = blockIdx.x * blockDim.x + threadIdx.x;
  if (idx >= K * N) return;
  int f = idx >> 9;
  int r = idx & 511;
  int lane = r >> 4;
  int j = r & 15;
  int ksTotal = K >> 5;
  int nt = f / ksTotal;
  int ks = f - nt * ksTotal;
  int k = ks * 32 + (lane >> 4) * 8 + (j & 7) + (j >> 3) * 16;
  int n = nt * 16 + (lane & 15);
  dst[idx] = (bf16_t)src[k * N + n];
}

// Stacked attention weights: rows [0,128)=Wr, [128,256)=Wqr, [256,384)=Ws
__global__ void swizzle3_kernel(const float* __restrict__ Wr,
                                const float* __restrict__ Wqr,
                                const float* __restrict__ Ws,
                                bf16_t* __restrict__ dst) {
  int idx = blockIdx.x * blockDim.x + threadIdx.x;
  if (idx >= 384 * 128) return;
  int f = idx >> 9;
  int r = idx & 511;
  int lane = r >> 4;
  int j = r & 15;
  int nt = f / 12;
  int ks = f - nt * 12;
  int k = ks * 32 + (lane >> 4) * 8 + (j & 7) + (j >> 3) * 16;
  int n = nt * 16 + (lane & 15);
  float v;
  if (k < 128)      v = Wr[k * 128 + n];
  else if (k < 256) v = Wqr[(k - 128) * 128 + n];
  else              v = Ws[(k - 256) * 128 + n];
  dst[idx] = (bf16_t)v;
}

__global__ void zero_kernel(float* __restrict__ p, size_t n) {
  size_t i = (size_t)blockIdx.x * blockDim.x + threadIdx.x;
  if (i < n) p[i] = 0.0f;
}

// ---------------------------------------------------------------------------
// Edge kernel: 4 waves / block, each wave owns 32 edges + PER_WAVE_LDS bytes.
// Pipeline per wave:
//   gather -> S1 fuse1 -> S2 fuse2(+temp_rel) -> S5 attn(alpha) ->
//   S3 gate(update, reset*hs) -> S4 trans(cand) -> message -> atomic scatter
// ---------------------------------------------------------------------------
__global__ void __launch_bounds__(128) edge_kernel(
    const int* __restrict__ edges, const int* __restrict__ q_rel,
    const float* __restrict__ hidden, const float* __restrict__ rela_embed,
    const float* __restrict__ pe_rel, const float* __restrict__ pe_abs,
    const int* __restrict__ ts_ptr, const float* __restrict__ fuse_b1,
    const float* __restrict__ fuse_b2, const float* __restrict__ gate_b,
    const float* __restrict__ trans_b, const float* __restrict__ wqr_b,
    const float* __restrict__ w_alpha, const bf16_t* __restrict__ W1s,
    const bf16_t* __restrict__ W2s, const bf16_t* __restrict__ Wgs,
    const bf16_t* __restrict__ Wts, const bf16_t* __restrict__ Was,
    float* __restrict__ agg, int E) {
  extern __shared__ char smem[];
  const int lane = threadIdx.x & 31;
  const int wid = threadIdx.x >> 5;
  char* wb = smem + wid * PER_WAVE_LDS;
  bf16_t* raw = (bf16_t*)(wb);             // 32 x 192 bf16  (temp_rel|pe|pe)
  bf16_t* agt = (bf16_t*)(wb + 12288);     // 32 x 384 bf16  (hr|h_qr|hs)
  bf16_t* rhs = (bf16_t*)(wb + 36864);     // 32 x 128 bf16  (x1, then reset*hs)
  float* hsf = (float*)(wb + 45056);       // 32 x 128 f32   hs
  float* updf = (float*)(wb + 61440);      // 32 x 128 f32   update gate
  float* alp = (float*)(wb + 77824);       // 32 f32         alpha accumulator

  const long gid = (long)blockIdx.x * 4 + wid;
  const long ebase = gid * 32;
  if (ebase >= (long)E) return;

  // warm the (large, later-used) gate weights in L2/WGP$
  __builtin_prefetch(Wgs + (size_t)threadIdx.x * 1536, 0, 3);

  const long e = ebase + lane;
  const bool valid = (e < (long)E);
  int r_idx = 0, relv = 0, subv = 0, objv = 0, tv = 0;
  if (valid) {
    const int* ep = edges + e * 7;
    r_idx = ep[0];
    relv = ep[2];
    subv = ep[4];
    objv = ep[5];
    tv = ep[6];
  }
  const int ts = ts_ptr[0];
  const int hqrow = q_rel[r_idx];
  const int trow = tv;
  const int arow = ts - tv;

  alp[lane] = 0.0f;

  // ---------------- gather (fully coalesced, one row per iteration) --------
  for (int m = 0; m < 32; ++m) {
    const int rm = __shfl(relv, m);
    const int sm_ = __shfl(subv, m);
    const int qm = __shfl(hqrow, m);
    const int trm = __shfl(trow, m);
    const int tam = __shfl(arow, m);

    const float4 re = ((const float4*)(rela_embed + (size_t)rm * 128))[lane];
    bf16_t* rr = raw + m * 192;
    *(v4bf*)(rr + lane * 4) = pack4(re.x, re.y, re.z, re.w);

    const float4 hh = ((const float4*)(hidden + (size_t)sm_ * 128))[lane];
    ((float4*)(hsf + m * 128))[lane] = hh;
    bf16_t* ga = agt + m * 384;
    *(v4bf*)(ga + 256 + lane * 4) = pack4(hh.x, hh.y, hh.z, hh.w);

    const float4 hq = ((const float4*)(rela_embed + (size_t)qm * 128))[lane];
    *(v4bf*)(ga + 128 + lane * 4) = pack4(hq.x, hq.y, hq.z, hq.w);

    rr[128 + lane] = (bf16_t)pe_rel[trm * 32 + lane];
    rr[160 + lane] = (bf16_t)pe_abs[tam * 32 + lane];
  }

  const int colb = lane & 15;
  const int mb = 8 * (lane >> 4);  // C-frag: lanes 0-15 -> M=r, 16-31 -> M=8+r

  // ---------------- S1: x1 = leaky(raw @ fuse_W1 + b1) ---------------------
#pragma unroll
  for (int nt = 0; nt < 8; ++nt) {
    v8f c0 = {}, c1 = {};
    gemm_acc<6>(raw, 192, W1s, nt * 6, lane, c0, c1);
    const int col = nt * 16 + colb;
    const float b = fuse_b1[col];
#pragma unroll
    for (int r = 0; r < 8; ++r) {
      rhs[(mb + r) * 128 + col] = (bf16_t)leaky(c0[r] + b);
      rhs[(mb + r + 16) * 128 + col] = (bf16_t)leaky(c1[r] + b);
    }
  }

  // ---------------- S2: hr = leaky(x1 @ fuse_W2 + b2) + temp_rel -----------
#pragma unroll
  for (int nt = 0; nt < 8; ++nt) {
    v8f c0 = {}, c1 = {};
    gemm_acc<4>(rhs, 128, W2s, nt * 4, lane, c0, c1);
    const int col = nt * 16 + colb;
    const float b = fuse_b2[col];
#pragma unroll
    for (int r = 0; r < 8; ++r) {
      {
        int m = mb + r;
        float tr = (float)raw[m * 192 + col];
        agt[m * 384 + col] = (bf16_t)(leaky(c0[r] + b) + tr);
      }
      {
        int m = mb + r + 16;
        float tr = (float)raw[m * 192 + col];
        agt[m * 384 + col] = (bf16_t)(leaky(c1[r] + b) + tr);
      }
    }
  }

  // ---------------- S5: alpha = leaky([hr|h_qr|hs] @ [Wr;Wqr;Ws] + b) @ w --
#pragma unroll
  for (int nt = 0; nt < 8; ++nt) {
    v8f c0 = {}, c1 = {};
    gemm_acc<12>(agt, 384, Was, nt * 12, lane, c0, c1);
    const int col = nt * 16 + colb;
    const float bq = wqr_b[col];
    const float wa = w_alpha[col];
#pragma unroll
    for (int r = 0; r < 8; ++r) {
      atomicAdd(&alp[mb + r], leaky(c0[r] + bq) * wa);
      atomicAdd(&alp[mb + r + 16], leaky(c1[r] + bq) * wa);
    }
  }
  {
    float av = alp[lane];
    alp[lane] = valid ? fast_sigmoid(av) : 0.0f;  // fold edge validity mask
  }

  // ---------------- S3: gates = sigmoid([hr|h_qr|hs] @ gate_W + b) ---------
#pragma unroll
  for (int nt = 0; nt < 16; ++nt) {
    v8f c0 = {}, c1 = {};
    gemm_acc<12>(agt, 384, Wgs, nt * 12, lane, c0, c1);
    const int col = nt * 16 + colb;
    const float b = gate_b[col];
#pragma unroll
    for (int r = 0; r < 8; ++r) {
      {
        int m = mb + r;
        float g = fast_sigmoid(c0[r] + b);
        if (nt < 8)
          updf[m * 128 + col] = g;  // update gate
        else {
          int c2 = col - 128;       // reset gate -> reset*hs (bf16, trans A)
          rhs[m * 128 + c2] = (bf16_t)(g * hsf[m * 128 + c2]);
        }
      }
      {
        int m = mb + r + 16;
        float g = fast_sigmoid(c1[r] + b);
        if (nt < 8)
          updf[m * 128 + col] = g;
        else {
          int c2 = col - 128;
          rhs[m * 128 + c2] = (bf16_t)(g * hsf[m * 128 + c2]);
        }
      }
    }
  }

  // ------- S4: cand = tanh([hr|reset*hs] @ trans_W + b); message; scatter --
#pragma unroll
  for (int nt = 0; nt < 8; ++nt) {
    v8f c0 = {}, c1 = {};
    gemm_acc<4>(agt, 384, Wts, nt * 8, lane, c0, c1);      // K 0..127  (hr)
    gemm_acc<4>(rhs, 128, Wts, nt * 8 + 4, lane, c0, c1);  // K 128..255
    const int col = nt * 16 + colb;
    const float b = trans_b[col];
#pragma unroll
    for (int r = 0; r < 8; ++r) {
      {
        int m = mb + r;
        float cand = fast_tanh(c0[r] + b);
        float u = updf[m * 128 + col];
        float hv = hsf[m * 128 + col];
        float msg = (1.0f - u) * hv + u * cand;
        float up = alp[m] * msg;  // sigmoid(alpha)*message (0 if invalid)
        int om = __shfl(objv, m);
        __hip_atomic_fetch_add(&agg[(size_t)om * 128 + col], up,
                               __ATOMIC_RELAXED, __HIP_MEMORY_SCOPE_AGENT);
      }
      {
        int m = mb + r + 16;
        float cand = fast_tanh(c1[r] + b);
        float u = updf[m * 128 + col];
        float hv = hsf[m * 128 + col];
        float msg = (1.0f - u) * hv + u * cand;
        float up = alp[m] * msg;
        int om = __shfl(objv, m);
        __hip_atomic_fetch_add(&agg[(size_t)om * 128 + col], up,
                               __ATOMIC_RELAXED, __HIP_MEMORY_SCOPE_AGENT);
      }
    }
  }
}

// ---------------------------------------------------------------------------
// out = agg @ W_h  (fp32; 0.6% of total FLOPs).  W_h (64KB) and each agg row
// are staged into LDS with CDNA5 async global->LDS copies (ASYNCcnt), no VGPR
// round-trip.
// ---------------------------------------------------------------------------
__global__ void __launch_bounds__(128) out_kernel(const float* __restrict__ agg,
                                                  const float* __restrict__ W_h,
                                                  float* __restrict__ out,
                                                  int n_node) {
  extern __shared__ char smem[];
  float* Wsh = (float*)smem;        // 128x128 f32
  float* rowb = Wsh + 16384;        // 128 f32
  // async-stage W_h: each thread copies 32 x 16B
#pragma unroll 4
  for (int i = threadIdx.x * 4; i < 16384; i += 512)
    async_copy_b128(Wsh + i, W_h + i);
  wait_async0();
  __syncthreads();

  int base = blockIdx.x * 16;
  for (int nd = base; nd < base + 16 && nd < n_node; ++nd) {
    if (threadIdx.x < 32)
      async_copy_b128(rowb + threadIdx.x * 4,
                      agg + (size_t)nd * 128 + threadIdx.x * 4);
    wait_async0();
    __syncthreads();
    float acc = 0.0f;
#pragma unroll 8
    for (int k = 0; k < 128; ++k) acc += rowb[k] * Wsh[k * 128 + threadIdx.x];
    out[(size_t)nd * 128 + threadIdx.x] = acc;
    __syncthreads();
  }
}

// ---------------------------------------------------------------------------
extern "C" void kernel_launch(void* const* d_in, const int* in_sizes, int n_in,
                              void* d_out, int out_size, void* d_ws,
                              size_t ws_size, hipStream_t stream) {
  const int* q_rel = (const int*)d_in[1];
  const float* hidden = (const float*)d_in[2];
  const int* edges = (const int*)d_in[3];
  const int* ts_ptr = (const int*)d_in[5];
  const float* rela = (const float*)d_in[6];
  const float* pe_rel = (const float*)d_in[7];
  const float* pe_abs = (const float*)d_in[8];
  const float* Ws = (const float*)d_in[9];
  const float* Wr = (const float*)d_in[10];
  const float* Wqr = (const float*)d_in[11];
  const float* wqr_b = (const float*)d_in[12];
  const float* fW1 = (const float*)d_in[13];
  const float* fb1 = (const float*)d_in[14];
  const float* fW2 = (const float*)d_in[15];
  const float* fb2 = (const float*)d_in[16];
  const float* w_alpha = (const float*)d_in[17];
  const float* gW = (const float*)d_in[18];
  const float* gb = (const float*)d_in[19];
  const float* tW = (const float*)d_in[20];
  const float* tb = (const float*)d_in[21];
  const float* W_h = (const float*)d_in[22];

  const int n_node = in_sizes[2] / 128;
  const int E = in_sizes[3] / 7;

  // workspace: [agg f32 n_node*128][bf16 fragment-swizzled weights]
  float* agg = (float*)d_ws;
  bf16_t* wbase = (bf16_t*)((char*)d_ws + (size_t)n_node * 128 * 4);
  bf16_t* W1s = wbase;                 // 192x128
  bf16_t* W2s = W1s + 192 * 128;       // 128x128
  bf16_t* Wgs = W2s + 128 * 128;       // 384x256
  bf16_t* Wts = Wgs + 384 * 256;       // 256x128
  bf16_t* Was = Wts + 256 * 128;       // 384x128 stacked [Wr;Wqr;Ws]

  swizzle_kernel<<<(192 * 128 + 255) / 256, 256, 0, stream>>>(fW1, W1s, 192, 128);
  swizzle_kernel<<<(128 * 128 + 255) / 256, 256, 0, stream>>>(fW2, W2s, 128, 128);
  swizzle_kernel<<<(384 * 256 + 255) / 256, 256, 0, stream>>>(gW, Wgs, 384, 256);
  swizzle_kernel<<<(256 * 128 + 255) / 256, 256, 0, stream>>>(tW, Wts, 256, 128);
  swizzle3_kernel<<<(384 * 128 + 255) / 256, 256, 0, stream>>>(Wr, Wqr, Ws, Was);

  size_t aggN = (size_t)n_node * 128;
  zero_kernel<<<(unsigned)((aggN + 255) / 256), 256, 0, stream>>>(agg, aggN);

  int nGroups = (E + 31) / 32;
  int nBlocks = (nGroups + 3) / 4;
  edge_kernel<<<nBlocks, 128, 4 * PER_WAVE_LDS, stream>>>(
      edges, q_rel, hidden, rela, pe_rel, pe_abs, ts_ptr, fb1, fb2, gb, tb,
      wqr_b, w_alpha, W1s, W2s, Wgs, Wts, Was, agg, E);

  out_kernel<<<(n_node + 15) / 16, 128, (16384 + 128) * 4, stream>>>(
      agg, W_h, (float*)d_out, n_node);
}
